// MultiHeadSelfAttention_60481729462285
// MI455X (gfx1250) — matrix operand order repaired and checked by reading, not compiled, not run on hardware
//
#include <hip/hip_runtime.h>

// ---------------------------------------------------------------------------
// MI455X (gfx1250) multi-head self-attention
//   bf16 WMMA (f32 accumulate) + Tensor Data Mover (TDM) double-buffered
//   LDS staging. All operands pre-arranged row-major so no in-kernel
//   transposes are needed: W matrices converted+transposed once, V written
//   transposed by the QKV epilogue.
// ---------------------------------------------------------------------------

typedef __bf16 bf16;
typedef __attribute__((ext_vector_type(16))) __bf16 v16bf;
typedef __attribute__((ext_vector_type(8)))  float   v8f;
typedef unsigned int       u32;
typedef unsigned long long u64;

union FragAB {
    v16bf v;
    uint4 q[2];
};

__device__ __forceinline__ v8f wmma_bf16f32(v16bf a, v16bf b, v8f c) {
    return __builtin_amdgcn_wmma_f32_16x16x32_bf16(false, a, false, b, (short)0, c, false, false);
}

#define DM   1024
#define SEQ  2048
#define NBAT 2
#define NH   16
#define HD   64
#define N3   3072

// ---------------------------------------------------------------------------
// Tensor Data Mover: 2D tile load, D# packed per CDNA5 ISA §8.3/8.4.
//   tile_d1 rows of tile_d0 bf16 elements, global row stride = stride_d0
//   elements, LDS rows padded by pad_amount+1 dwords every 2^(pad_interval+1)
//   dwords (i.e. per row).
// ---------------------------------------------------------------------------
#if defined(__AMDGCN__) && __has_builtin(__builtin_amdgcn_tensor_load_to_lds)
#define USE_TDM 1
#else
#define USE_TDM 0
#endif

#if USE_TDM
typedef __attribute__((ext_vector_type(4))) u32 tdm_v4u;
typedef __attribute__((ext_vector_type(8))) int tdm_v8i;
typedef __attribute__((ext_vector_type(4))) int tdm_v4i;

__device__ __forceinline__ void tdm_issue_2d(const void* g, u32 lds_off,
                                             u32 tile_d0, u32 tile_d1,
                                             u32 stride_d0,
                                             u32 pad_interval, u32 pad_amount)
{
    const u64 ga = (u64)g;
    tdm_v4u g0;
    g0.x = 1u;                                             // count=1 (valid), user mode
    g0.y = lds_off;                                        // lds_addr [63:32]
    g0.z = (u32)ga;                                        // global_addr lo
    g0.w = ((u32)(ga >> 32) & 0x01FFFFFFu) | (2u << 30);   // global_addr[56:32] | type=2
    tdm_v8i g1;
    g1[0] = (int)((1u << 16)                               // data_size = 2 bytes
                | (1u << 20)                               // pad_enable
                | (pad_interval << 22) | (pad_amount << 25));
    g1[1] = (int)((tile_d0 & 0xFFFFu) << 16);              // tensor_dim0[15:0] (== tile_d0)
    g1[2] = (int)(((tile_d0 >> 16) & 0xFFFFu) | ((tile_d1 & 0xFFFFu) << 16)); // tensor_dim1
    g1[3] = (int)(((tile_d1 >> 16) & 0xFFFFu) | ((tile_d0 & 0xFFFFu) << 16)); // tile_dim0
    g1[4] = (int)(tile_d1 & 0xFFFFu);                      // tile_dim1 (tile_dim2 = 0)
    g1[5] = (int)stride_d0;                                // tensor_dim0_stride[31:0]
    g1[6] = 0;                                             // stride hi / dim1_stride lo
    g1[7] = 0;
    tdm_v4i z4 = {0, 0, 0, 0};                             // 2D: groups 2/3 unused
    tdm_v8i z8 = {0, 0, 0, 0, 0, 0, 0, 0};
    // amdgpu-toolchain (clang-23 / therock-10.0) 6-arg form
    __builtin_amdgcn_tensor_load_to_lds(g0, g1, z4, z4, z8, 0);
}
#endif

// ---------------------------------------------------------------------------
// fp32 -> bf16 conversion (vectorized x4)
// ---------------------------------------------------------------------------
__global__ void mhsa_cvt_bf16(const float* __restrict__ in, bf16* __restrict__ out, int n4) {
    int i = blockIdx.x * blockDim.x + threadIdx.x;
    if (i >= n4) return;
    float4 f = reinterpret_cast<const float4*>(in)[i];
    union { bf16 b[4]; uint2 u; } pk;
    pk.b[0] = (bf16)f.x; pk.b[1] = (bf16)f.y; pk.b[2] = (bf16)f.z; pk.b[3] = (bf16)f.w;
    reinterpret_cast<uint2*>(out)[i] = pk.u;
}

// fp32 [R][C] -> bf16 transposed [C][R]
__global__ void mhsa_cvt_t_bf16(const float* __restrict__ in, bf16* __restrict__ out,
                                int R, int C) {
    int i = blockIdx.x * blockDim.x + threadIdx.x;
    if (i >= R * C) return;
    int r = i / C, c = i - r * C;
    out[(size_t)c * R + r] = (bf16)in[i];
}

// ---------------------------------------------------------------------------
// Shared GEMM mainloop: C[128x128] += A[128xK] * Bt[N=128 rows][K], bf16 in,
// f32 acc. 256 threads = 8 waves (4M x 2N); each wave owns 2x4 16x16 tiles.
// LDS per buffer: A [128][40], Bt [128][40]; double buffered, TDM staged.
// ---------------------------------------------------------------------------
__device__ __forceinline__ void gemm_bf16_tile(
    const bf16* __restrict__ A, int lda,
    const bf16* __restrict__ Bt, int ldb,
    int K, int m0, int n0,
    bf16* sA, bf16* sBt,                 // each 2 * 128*40 elements
    v8f acc[2][4])
{
    const int t     = threadIdx.x;
    const int lane  = t & 31;
    const int wave  = t >> 5;
    const int waveM = wave >> 1;
    const int waveN = wave & 1;
    const int hf    = lane >> 4;
    const int l16   = lane & 15;

#pragma unroll
    for (int mt = 0; mt < 2; mt++)
#pragma unroll
        for (int nt = 0; nt < 4; nt++) acc[mt][nt] = {};

    const int NI = K / 32;

    auto stage = [&](int ki) {
        const int bsel = ki & 1;
        bf16* dA = sA  + bsel * (128 * 40);
        bf16* dB = sBt + bsel * (128 * 40);
        const int k0 = ki * 32;
#if USE_TDM
        if (wave == 0) {
            // rows of 32 bf16 = 16 dwords -> pad_interval=3 (16 dw), +4 dw -> stride 40
            tdm_issue_2d(A  + (size_t)m0 * lda + k0, (u32)(size_t)dA, 32, 128, (u32)lda, 3, 3);
            tdm_issue_2d(Bt + (size_t)n0 * ldb + k0, (u32)(size_t)dB, 32, 128, (u32)ldb, 3, 3);
        }
#else
        const int r = t >> 2;          // 0..63, two row passes
        const int c = (t & 3) * 8;
        *reinterpret_cast<uint4*>(dA + r * 40 + c) =
            *reinterpret_cast<const uint4*>(A + (size_t)(m0 + r) * lda + k0 + c);
        *reinterpret_cast<uint4*>(dA + (r + 64) * 40 + c) =
            *reinterpret_cast<const uint4*>(A + (size_t)(m0 + r + 64) * lda + k0 + c);
        *reinterpret_cast<uint4*>(dB + r * 40 + c) =
            *reinterpret_cast<const uint4*>(Bt + (size_t)(n0 + r) * ldb + k0 + c);
        *reinterpret_cast<uint4*>(dB + (r + 64) * 40 + c) =
            *reinterpret_cast<const uint4*>(Bt + (size_t)(n0 + r + 64) * ldb + k0 + c);
#endif
    };

    stage(0);
    for (int ki = 0; ki < NI; ki++) {
        if (ki + 1 < NI) stage(ki + 1);          // prefetch next tile (DMA overlaps WMMA)
#if USE_TDM
        if (wave == 0) {
            if (ki + 1 < NI) __builtin_amdgcn_s_wait_tensorcnt(2);  // current tile landed
            else             __builtin_amdgcn_s_wait_tensorcnt(0);
        }
#endif
        __syncthreads();

        const bf16* cA = sA  + (ki & 1) * (128 * 40);
        const bf16* cB = sBt + (ki & 1) * (128 * 40);
        const int kbase = hf ? 8 : 0;    // A frag: lanes 16-31 hold K=8..15,24..31
        const int koff  = hf ? 8 : 0;    // B frag: lanes 16-31 hold K=8..15,24..31
        FragAB afr[2];
#pragma unroll
        for (int mt = 0; mt < 2; mt++) {
            const bf16* base = cA + (waveM * 32 + mt * 16 + l16) * 40 + kbase;
            afr[mt].q[0] = *reinterpret_cast<const uint4*>(base);
            afr[mt].q[1] = *reinterpret_cast<const uint4*>(base + 16);
        }
#pragma unroll
        for (int nt = 0; nt < 4; nt++) {
            FragAB bfr;
            const bf16* base = cB + (waveN * 64 + nt * 16 + l16) * 40 + koff;
            bfr.q[0] = *reinterpret_cast<const uint4*>(base);
            bfr.q[1] = *reinterpret_cast<const uint4*>(base + 16);
#pragma unroll
            for (int mt = 0; mt < 2; mt++)
                acc[mt][nt] = wmma_bf16f32(afr[mt].v, bfr.v, acc[mt][nt]);
        }
        __syncthreads();
    }
}

// ---------------------------------------------------------------------------
// GEMM 1: qkv = x @ W_qkv + b_qkv; scatter Q (scaled), K into [B,H,S,d] and
// V transposed into [B,H,d,S] (so attention can TDM-stage it row-major).
// ---------------------------------------------------------------------------
__global__ __launch_bounds__(256) void mhsa_qkv_gemm(
    const bf16* __restrict__ xb, const bf16* __restrict__ Wt,
    const float* __restrict__ bias,
    bf16* __restrict__ Qb, bf16* __restrict__ Kb, bf16* __restrict__ Vtb)
{
    __shared__ bf16 sA[2 * 128 * 40];
    __shared__ bf16 sBt[2 * 128 * 40];
    v8f acc[2][4];
    const int m0 = blockIdx.x * 128;
    const int n0 = blockIdx.y * 128;
    gemm_bf16_tile(xb, DM, Wt, DM, DM, m0, n0, sA, sBt, acc);

    const int t = threadIdx.x, lane = t & 31, wave = t >> 5;
    const int waveM = wave >> 1, waveN = wave & 1;
    const int hf = lane >> 4, l16 = lane & 15;

#pragma unroll
    for (int mt = 0; mt < 2; mt++) {
        const int mbase = m0 + waveM * 32 + mt * 16 + hf * 8;
#pragma unroll
        for (int nt = 0; nt < 4; nt++) {
            const int n = n0 + waveN * 64 + nt * 16 + l16;
            const int h = n / 192;
            const int j = n - h * 192;       // 0..63 Q, 64..127 K, 128..191 V
            const float bv = bias[n];
#pragma unroll
            for (int r = 0; r < 8; r++) {
                const int m = mbase + r;
                const int b = m >> 11;
                const int s = m & 2047;
                const float val = acc[mt][nt][r] + bv;
                const size_t bhh = (size_t)(b * NH + h);
                if (j < 64)       Qb[(bhh * SEQ + s) * HD + j] = (bf16)(val * 0.125f);
                else if (j < 128) Kb[(bhh * SEQ + s) * HD + j - 64] = (bf16)val;
                else              Vtb[(bhh * HD + (j - 128)) * SEQ + s] = (bf16)val;
            }
        }
    }
}

// ---------------------------------------------------------------------------
// GEMM 2: out = attn_flat @ W_out^T(pre-transposed) + b_out  (fp32 output)
// ---------------------------------------------------------------------------
__global__ __launch_bounds__(256) void mhsa_out_gemm(
    const bf16* __restrict__ Ab, const bf16* __restrict__ Wt,
    const float* __restrict__ bias, float* __restrict__ out)
{
    __shared__ bf16 sA[2 * 128 * 40];
    __shared__ bf16 sBt[2 * 128 * 40];
    v8f acc[2][4];
    const int m0 = blockIdx.x * 128;
    const int n0 = blockIdx.y * 128;
    gemm_bf16_tile(Ab, DM, Wt, DM, DM, m0, n0, sA, sBt, acc);

    const int t = threadIdx.x, lane = t & 31, wave = t >> 5;
    const int waveM = wave >> 1, waveN = wave & 1;
    const int hf = lane >> 4, l16 = lane & 15;

#pragma unroll
    for (int mt = 0; mt < 2; mt++) {
        const int mbase = m0 + waveM * 32 + mt * 16 + hf * 8;
#pragma unroll
        for (int nt = 0; nt < 4; nt++) {
            const int n = n0 + waveN * 64 + nt * 16 + l16;
            const float bv = bias[n];
#pragma unroll
            for (int r = 0; r < 8; r++)
                out[(size_t)(mbase + r) * DM + n] = acc[mt][nt][r] + bv;
        }
    }
}

// ---------------------------------------------------------------------------
// Flash attention: one block per (b,h, 128-query tile); 8 waves x 16 queries.
// K tiles [64 keys][64 dims] and V^T tiles [64 dims][64 keys] TDM-staged,
// double buffered; online softmax; P routed through per-wave LDS to convert
// C-layout -> A-layout for the P@V WMMAs.
// ---------------------------------------------------------------------------
__global__ __launch_bounds__(256) void mhsa_flash_attn(
    const bf16* __restrict__ Qb, const bf16* __restrict__ Kb,
    const bf16* __restrict__ Vtb, bf16* __restrict__ Ob)
{
    __shared__ bf16 sK[2 * 64 * 72];     // [key][dim]  (B-frag layout for Q@K^T)
    __shared__ bf16 sVt[2 * 64 * 72];    // [dim][key]  (B-frag layout for P@V)
    __shared__ bf16 sP[8 * 16 * 72];     // per-wave P tile [16][64] (+pad)

    const int bh = blockIdx.y;
    const int q0 = blockIdx.x * 128;
    const bf16* Qp  = Qb  + (size_t)bh * SEQ * HD;
    const bf16* Kp  = Kb  + (size_t)bh * SEQ * HD;
    const bf16* Vtp = Vtb + (size_t)bh * HD * SEQ;

    const int t = threadIdx.x, lane = t & 31, wave = t >> 5;
    const int hf = lane >> 4, l16 = lane & 15;
    const int kbase = hf ? 8 : 0;
    const int koff  = hf ? 8 : 0;

    // Q fragments (2 k-chunks of 32) cached in VGPRs for the whole kv loop.
    FragAB qf[2];
    {
        const bf16* qrow = Qp + (size_t)(q0 + wave * 16 + l16) * HD;
#pragma unroll
        for (int kc = 0; kc < 2; kc++) {
            qf[kc].q[0] = *reinterpret_cast<const uint4*>(qrow + kc * 32 + kbase);
            qf[kc].q[1] = *reinterpret_cast<const uint4*>(qrow + kc * 32 + kbase + 16);
        }
    }

    v8f o[4];
#pragma unroll
    for (int i = 0; i < 4; i++) o[i] = {};
    float mrow[8], lrow[8];
#pragma unroll
    for (int r = 0; r < 8; r++) { mrow[r] = -1e30f; lrow[r] = 0.0f; }

    auto stageKV = [&](int ki) {
        const int bsel = ki & 1;
        bf16* dK = sK  + bsel * (64 * 72);
        bf16* dV = sVt + bsel * (64 * 72);
        const int kv0 = ki * 64;
#if USE_TDM
        if (wave == 0) {
            // rows of 64 bf16 = 32 dwords -> pad_interval=4, +4 dwords -> stride 72
            tdm_issue_2d(Kp + (size_t)kv0 * HD, (u32)(size_t)dK, 64, 64, HD, 4, 3);
            tdm_issue_2d(Vtp + kv0,            (u32)(size_t)dV, 64, 64, SEQ, 4, 3);
        }
#else
        const int r = t >> 2;
        const int c = (t & 3) * 16;
        const bf16* ks = Kp + (size_t)(kv0 + r) * HD + c;
        *reinterpret_cast<uint4*>(dK + r * 72 + c)     = *reinterpret_cast<const uint4*>(ks);
        *reinterpret_cast<uint4*>(dK + r * 72 + c + 8) = *reinterpret_cast<const uint4*>(ks + 8);
        const bf16* vs = Vtp + (size_t)r * SEQ + kv0 + c;
        *reinterpret_cast<uint4*>(dV + r * 72 + c)     = *reinterpret_cast<const uint4*>(vs);
        *reinterpret_cast<uint4*>(dV + r * 72 + c + 8) = *reinterpret_cast<const uint4*>(vs + 8);
#endif
    };

    const int NKV = SEQ / 64;
    stageKV(0);
    for (int ki = 0; ki < NKV; ki++) {
        if (ki + 1 < NKV) stageKV(ki + 1);
#if USE_TDM
        if (wave == 0) {
            if (ki + 1 < NKV) __builtin_amdgcn_s_wait_tensorcnt(2);
            else              __builtin_amdgcn_s_wait_tensorcnt(0);
        }
#endif
        __syncthreads();

        const bf16* cK = sK  + (ki & 1) * (64 * 72);
        const bf16* cV = sVt + (ki & 1) * (64 * 72);

        // ---- scores: S = Q * K^T  (Q pre-scaled by 1/sqrt(d)) ----
        v8f sc[4];
#pragma unroll
        for (int i = 0; i < 4; i++) sc[i] = {};
#pragma unroll
        for (int nt = 0; nt < 4; nt++) {
#pragma unroll
            for (int kc = 0; kc < 2; kc++) {
                FragAB kf;
                const bf16* base = cK + (nt * 16 + l16) * 72 + kc * 32 + koff;
                kf.q[0] = *reinterpret_cast<const uint4*>(base);
                kf.q[1] = *reinterpret_cast<const uint4*>(base + 16);
                sc[nt] = wmma_bf16f32(qf[kc].v, kf.v, sc[nt]);
            }
        }

        // ---- online softmax (rows live in 16-lane halves per C layout) ----
        float alpha[8];
#pragma unroll
        for (int r = 0; r < 8; r++) {
            float mx = sc[0][r];
#pragma unroll
            for (int nt = 1; nt < 4; nt++) mx = fmaxf(mx, sc[nt][r]);
            mx = fmaxf(mx, __shfl_xor(mx, 1));
            mx = fmaxf(mx, __shfl_xor(mx, 2));
            mx = fmaxf(mx, __shfl_xor(mx, 4));
            mx = fmaxf(mx, __shfl_xor(mx, 8));
            const float mnew = fmaxf(mrow[r], mx);
            alpha[r] = __expf(mrow[r] - mnew);
            mrow[r]  = mnew;
        }

        bf16* pbase = sP + wave * 16 * 72;
#pragma unroll
        for (int r = 0; r < 8; r++) {
            float rs = 0.0f;
#pragma unroll
            for (int nt = 0; nt < 4; nt++) {
                const float p = __expf(sc[nt][r] - mrow[r]);
                rs += p;
                pbase[(r + hf * 8) * 72 + nt * 16 + l16] = (bf16)p;  // C-layout scatter
            }
            rs += __shfl_xor(rs, 1);
            rs += __shfl_xor(rs, 2);
            rs += __shfl_xor(rs, 4);
            rs += __shfl_xor(rs, 8);
            lrow[r] = lrow[r] * alpha[r] + rs;
        }
#pragma unroll
        for (int nt = 0; nt < 4; nt++)
#pragma unroll
            for (int r = 0; r < 8; r++)
                o[nt][r] *= alpha[r];

        // ---- O += P @ V  (same-wave LDS round trip is in-order) ----
#pragma unroll
        for (int kc = 0; kc < 2; kc++) {
            FragAB pf;
            const bf16* pb = pbase + l16 * 72 + kc * 32 + kbase;
            pf.q[0] = *reinterpret_cast<const uint4*>(pb);
            pf.q[1] = *reinterpret_cast<const uint4*>(pb + 16);
#pragma unroll
            for (int nt = 0; nt < 4; nt++) {
                FragAB vf;
                const bf16* vb = cV + (nt * 16 + l16) * 72 + kc * 32 + koff;
                vf.q[0] = *reinterpret_cast<const uint4*>(vb);
                vf.q[1] = *reinterpret_cast<const uint4*>(vb + 16);
                o[nt] = wmma_bf16f32(pf.v, vf.v, o[nt]);
            }
        }
        __syncthreads();
    }

    // ---- epilogue: normalize, store bf16 into [B*S, D] for out-projection ----
    const int h = bh & (NH - 1);
    const int b = bh >> 4;
#pragma unroll
    for (int nt = 0; nt < 4; nt++) {
#pragma unroll
        for (int r = 0; r < 8; r++) {
            const int m   = q0 + wave * 16 + hf * 8 + r;
            const int col = h * HD + nt * 16 + l16;
            Ob[(size_t)(b * SEQ + m) * DM + col] = (bf16)(o[nt][r] / lrow[r]);
        }
    }
}

// ---------------------------------------------------------------------------
// Launch
// ---------------------------------------------------------------------------
extern "C" void kernel_launch(void* const* d_in, const int* in_sizes, int n_in,
                              void* d_out, int out_size, void* d_ws, size_t ws_size,
                              hipStream_t stream) {
    (void)in_sizes; (void)n_in; (void)out_size; (void)ws_size;
    const float* x     = (const float*)d_in[0];
    const float* W_qkv = (const float*)d_in[1];
    const float* b_qkv = (const float*)d_in[2];
    const float* W_out = (const float*)d_in[3];
    const float* b_out = (const float*)d_in[4];
    float* out = (float*)d_out;

    // Workspace layout (bf16 elements), ~48 MB total
    bf16* xb     = (bf16*)d_ws;
    bf16* Wqkv_t = xb     + (size_t)NBAT * SEQ * DM;   // [3072][1024]
    bf16* Wout_t = Wqkv_t + (size_t)DM * N3;           // [1024][1024]
    bf16* Qb     = Wout_t + (size_t)DM * DM;           // [B,H,S,d]
    bf16* Kb     = Qb     + (size_t)NBAT * NH * SEQ * HD;
    bf16* Vtb    = Kb     + (size_t)NBAT * NH * SEQ * HD;  // [B,H,d,S]
    bf16* Ob     = Vtb    + (size_t)NBAT * NH * SEQ * HD;  // [B*S, D]

    // 1) fp32 -> bf16 conversions (weights transposed so GEMM B is row-major [N][K])
    {
        int n4 = NBAT * SEQ * DM / 4;
        mhsa_cvt_bf16<<<(n4 + 255) / 256, 256, 0, stream>>>(x, xb, n4);
        int n = DM * N3;
        mhsa_cvt_t_bf16<<<(n + 255) / 256, 256, 0, stream>>>(W_qkv, Wqkv_t, DM, N3);
        n = DM * DM;
        mhsa_cvt_t_bf16<<<(n + 255) / 256, 256, 0, stream>>>(W_out, Wout_t, DM, DM);
    }

    // 2) QKV projection (4096 x 3072 x 1024)
    {
        dim3 grid(NBAT * SEQ / 128, N3 / 128);
        mhsa_qkv_gemm<<<grid, 256, 0, stream>>>(xb, Wqkv_t, b_qkv, Qb, Kb, Vtb);
    }

    // 3) Flash attention (per b,h, 128-query tile)
    {
        dim3 grid(SEQ / 128, NBAT * NH);
        mhsa_flash_attn<<<grid, 256, 0, stream>>>(Qb, Kb, Vtb, Ob);
    }

    // 4) Output projection (4096 x 1024 x 1024) + bias -> fp32
    {
        dim3 grid(NBAT * SEQ / 128, DM / 128);
        mhsa_out_gemm<<<grid, 256, 0, stream>>>(Ob, Wout_t, b_out, out);
    }
}
